// minLSTM_25907242730139
// MI455X (gfx1250) — matrix operand order repaired and checked by reading, not compiled, not run on hardware
//
#include <hip/hip_runtime.h>
#include <hip/hip_bf16.h>

typedef __attribute__((ext_vector_type(16))) __bf16 v16bf;
typedef __attribute__((ext_vector_type(8)))  __bf16 v8bf;
typedef __attribute__((ext_vector_type(8)))  float  v8f;
typedef __attribute__((ext_vector_type(4)))  unsigned int u32x4;
typedef __attribute__((ext_vector_type(8)))  int i32x8;
typedef __attribute__((ext_vector_type(4)))  int i32x4;

#define BB    8
#define TT    4096
#define DD    1024
#define BT    (BB * TT)          // 32768 rows
#define KSTEP 64
#define LDST  72                 // LDS row stride in bf16 elems: 64 + 8 pad (144B, 16B-aligned)

// ---------------- fast math helpers (hardware v_exp_f32 / v_log_f32) ----------------

__device__ __forceinline__ float softplus0_f(float z) {
    // softplus(z) for z <= 0: log(1 + exp(z)); exp(z) in [0,1] -> fast hw log is fine
    return __logf(1.0f + __expf(z));
}

__device__ __forceinline__ float log_sigmoid_f(float a) {
    // -softplus(-a) = min(a,0) - softplus(-|a|)
    return fminf(a, 0.0f) - softplus0_f(-fabsf(a));
}

__device__ __forceinline__ float log_g_f(float h) {
    // h>=0: log(h+0.5); h<0: -softplus(-h) = h - softplus(h)  (h<=0 so softplus0 ok)
    return (h >= 0.0f) ? __logf(h + 0.5f) : (h - softplus0_f(h));
}

__device__ __forceinline__ float logaddexp_f(float a, float b) {
    float m = fmaxf(a, b);
    return m + softplus0_f(-fabsf(a - b));   // -inf inputs: exp(-inf)=0 -> returns m
}

// ---------------- stage 1: fp32 -> bf16 convert ----------------

__global__ __launch_bounds__(256) void cvt_f32_bf16(const float* __restrict__ src,
                                                    __bf16* __restrict__ dst, int n8) {
    int i = blockIdx.x * blockDim.x + threadIdx.x;
    if (i >= n8) return;
    const float4* s4 = (const float4*)src;
    float4 a = s4[2 * i];
    float4 b = s4[2 * i + 1];
    v8bf o;
    o[0] = (__bf16)a.x; o[1] = (__bf16)a.y; o[2] = (__bf16)a.z; o[3] = (__bf16)a.w;
    o[4] = (__bf16)b.x; o[5] = (__bf16)b.y; o[6] = (__bf16)b.z; o[7] = (__bf16)b.w;
    ((v8bf*)dst)[i] = o;
}

// ---------------- TDM descriptor build + issue (2D tile, bf16, row-padded LDS) ----------------
// D# layout per CDNA5 ISA. All inputs are wave-uniform.

__device__ __forceinline__ void tdm_load_2d_bf16(unsigned int lds_byte_addr,
                                                 const void* gaddr,
                                                 unsigned int tensor_d0,   // inner len (elems)
                                                 unsigned int tensor_d1,   // #rows
                                                 unsigned int tile_d0,     // tile inner (elems)
                                                 unsigned int tile_d1,     // tile rows
                                                 unsigned int stride_d0) { // row stride (elems)
    unsigned long long ga = (unsigned long long)(uintptr_t)gaddr;
    u32x4 g0;
    g0[0] = 1u;                                            // count=1, user descriptor
    g0[1] = lds_byte_addr;                                 // lds_addr [63:32]
    g0[2] = (unsigned int)(ga & 0xFFFFFFFFu);              // global_addr[31:0]
    g0[3] = (unsigned int)((ga >> 32) & 0x01FFFFFFu)       // global_addr[56:32]
            | 0x80000000u;                                 // type=2 ("image") bits 127:126

    // pad: rows of 64 bf16 = 128B = 32 DWORDs -> interval code 4; pad 16B = 4 DWORDs -> amount code 3
    i32x8 g1;
    g1[0] = (int)((1u << 16)            // data_size = 1 -> 2 bytes
                  | (1u << 20)          // pad_enable
                  | (4u << 22)          // pad_interval = 32 DWORDs
                  | (3u << 25));        // pad_amount   = 4 DWORDs
    g1[1] = (int)(tensor_d0 << 16);                        // tensor_dim0[15:0] @ bits 63:48
    g1[2] = (int)((tensor_d0 >> 16) | (tensor_d1 << 16));  // dim0 hi | dim1 lo
    g1[3] = (int)((tensor_d1 >> 16) | (tile_d0 << 16));    // dim1 hi | tile_dim0
    g1[4] = (int)(tile_d1 & 0xFFFFu);                      // tile_dim1 (tile_dim2 = 0)
    g1[5] = (int)stride_d0;                                // tensor_dim0_stride[31:0]
    g1[6] = 0;                                             // stride hi / dim1_stride lo
    g1[7] = 0;
    i32x4 gz = {0, 0, 0, 0};
#if __has_include(<hip/amd_detail/amd_gfx1250_TDM.h>)
    i32x8 z8 = {0, 0, 0, 0, 0, 0, 0, 0};
    __builtin_amdgcn_tensor_load_to_lds(g0, g1, gz, gz, z8, 0);
#else
    __builtin_amdgcn_tensor_load_to_lds(g0, g1, gz, gz, 0);
#endif
}

// ---------------- stage 2: fused 3-way gates GEMM + activation epilogue ----------------
// out[b,t,e] = sum_d x[b,t,d] * W[e,d].  A = x (M x K), B = W^T: B column n = W row n.
// Workgroup: 8 waves, tile M=128 x N=64, K-step 64, TDM double-buffered staging.

#define A_TILE_B  (128 * LDST * 2)          // 18432 B per buffer
#define B_TILE_B  (64 * LDST * 2)           // per gate: 9216 B

__global__ __launch_bounds__(256) void gates_gemm(const __bf16* __restrict__ xbf,   // [BT, D]
                                                  const __bf16* __restrict__ wbf,   // [3, D, D] (f,i,h)
                                                  float* __restrict__ lfOut,        // [BT, D]
                                                  float* __restrict__ lvOut) {      // [BT, D] (== d_out)
    __shared__ __bf16 As[2 * 128 * LDST];       // 36864 B
    __shared__ __bf16 Bs[2 * 3 * 64 * LDST];    // 55296 B

    const int tid  = threadIdx.x;
    const int lane = tid & 31;
    const int w    = tid >> 5;
    const int wm   = w >> 1;                 // 0..3
    const int wn   = w & 1;                  // 0..1
    const int nBase = blockIdx.x * 64;
    const int mBase = blockIdx.y * 128;

    const unsigned int asBase = (unsigned int)(uintptr_t)(&As[0]);
    const unsigned int bsBase = (unsigned int)(uintptr_t)(&Bs[0]);

    v8f acc[3][2][2] = {};

    const int alo = (lane < 16) ? 0 : 8;     // A per-lane K base (ISA 16-bit A layout)
    const int blo = (lane < 16) ? 0 : 8;     // B per-lane K base (same K order as A)

    // prologue: DMA buffer 0
    if (w == 0) {
        tdm_load_2d_bf16(asBase, xbf + (size_t)mBase * DD, DD, BT, KSTEP, 128, DD);
        #pragma unroll
        for (int g = 0; g < 3; ++g)
            tdm_load_2d_bf16(bsBase + g * B_TILE_B,
                             wbf + ((size_t)g * DD + nBase) * DD, DD, DD, KSTEP, 64, DD);
    }

    int buf = 0;
    for (int k0 = 0; k0 < DD; k0 += KSTEP) {
        if (w == 0) {
            if (k0 + KSTEP < DD) {
                const int nb = buf ^ 1;
                tdm_load_2d_bf16(asBase + nb * A_TILE_B,
                                 xbf + (size_t)mBase * DD + (k0 + KSTEP), DD, BT, KSTEP, 128, DD);
                #pragma unroll
                for (int g = 0; g < 3; ++g)
                    tdm_load_2d_bf16(bsBase + nb * (3 * B_TILE_B) + g * B_TILE_B,
                                     wbf + ((size_t)g * DD + nBase) * DD + (k0 + KSTEP),
                                     DD, DD, KSTEP, 64, DD);
                __builtin_amdgcn_s_wait_tensorcnt(4);   // buffer `buf` complete, 4 prefetch in flight
            } else {
                __builtin_amdgcn_s_wait_tensorcnt(0);
            }
        }
        __syncthreads();

        const __bf16* Ab = As + buf * (128 * LDST);
        const __bf16* Bb = Bs + buf * (3 * 64 * LDST);

        #pragma unroll
        for (int ks = 0; ks < KSTEP; ks += 32) {
            // A fragments: lane = M row; VGPR0..3 <- K alo..alo+7, VGPR4..7 <- K alo+16..alo+23
            v16bf afrag[2];
            #pragma unroll
            for (int mi = 0; mi < 2; ++mi) {
                const int row = wm * 32 + mi * 16 + (lane & 15);
                v8bf lo = *(const v8bf*)(Ab + row * LDST + ks + alo);
                v8bf hi = *(const v8bf*)(Ab + row * LDST + ks + alo + 16);
                afrag[mi] = __builtin_shufflevector(lo, hi, 0, 1, 2, 3, 4, 5, 6, 7,
                                                            8, 9, 10, 11, 12, 13, 14, 15);
            }
            // B fragments: lane = N column; K blo..blo+7, blo+16..blo+23 (matches A)
            #pragma unroll
            for (int g = 0; g < 3; ++g) {
                #pragma unroll
                for (int ni = 0; ni < 2; ++ni) {
                    const int e = wn * 32 + ni * 16 + (lane & 15);
                    v8bf lo = *(const v8bf*)(Bb + (g * 64 + e) * LDST + ks + blo);
                    v8bf hi = *(const v8bf*)(Bb + (g * 64 + e) * LDST + ks + blo + 16);
                    v16bf bfrag = __builtin_shufflevector(lo, hi, 0, 1, 2, 3, 4, 5, 6, 7,
                                                                  8, 9, 10, 11, 12, 13, 14, 15);
                    #pragma unroll
                    for (int mi = 0; mi < 2; ++mi) {
                        acc[g][mi][ni] = __builtin_amdgcn_wmma_f32_16x16x32_bf16(
                            false, afrag[mi], false, bfrag, (short)0, acc[g][mi][ni], false, false);
                    }
                }
            }
        }
        __syncthreads();
        buf ^= 1;
    }

    // epilogue: C layout: VGPR r, lanes 0-15 -> M=r, lanes 16-31 -> M=r+8; N = lane%16
    #pragma unroll
    for (int mi = 0; mi < 2; ++mi) {
        #pragma unroll
        for (int ni = 0; ni < 2; ++ni) {
            const int gcol  = nBase + wn * 32 + ni * 16 + (lane & 15);
            const int rbase = mBase + wm * 32 + mi * 16 + ((lane >= 16) ? 8 : 0);
            #pragma unroll
            for (int r = 0; r < 8; ++r) {
                const size_t idx = (size_t)(rbase + r) * DD + gcol;
                float af = acc[0][mi][ni][r];
                float ai = acc[1][mi][ni][r];
                float ah = acc[2][mi][ni][r];
                lfOut[idx] = log_sigmoid_f(af);
                lvOut[idx] = log_sigmoid_f(ai) + log_g_f(ah);
            }
        }
    }
}

// ---------------- stage 3: chunked associative scan over T ----------------
// h_t = f_t * h_{t-1} + exp(lv_t) in log space; chunk map (la, lb) composed onto state:
// h' = logaddexp(la + h, lb). Block: 32 lanes over d (coalesced), 8 chunks of 512 steps.

__global__ __launch_bounds__(256) void scan_kernel(const float* __restrict__ lf,
                                                   float* __restrict__ lv_out) {
    __shared__ float sLa[8][32];
    __shared__ float sLb[8][32];

    const int lx = threadIdx.x;                 // 0..31 (d within group)
    const int cy = threadIdx.y;                 // chunk 0..7
    const int b  = blockIdx.x >> 5;             // 0..7
    const int d  = ((blockIdx.x & 31) << 5) + lx;
    const int CL = TT / 8;                      // 512
    const int t0 = cy * CL;
    const size_t chanBase = (size_t)b * TT * DD + d;

    // pass 1: per-chunk summary (la = sum log_f, lb via recurrence from -inf)
    float la = 0.0f;
    float lb = -INFINITY;
    for (int t = t0; t < t0 + CL; ++t) {
        const size_t idx = chanBase + (size_t)t * DD;
        float f = lf[idx];
        float v = lv_out[idx];
        la += f;
        lb = logaddexp_f(lb + f, v);
    }
    sLa[cy][lx] = la;
    sLb[cy][lx] = lb;
    __syncthreads();

    // exclusive prefix: start state = log(0.5), fold lower chunks' maps
    float h = -0.6931471805599453f;             // log(0.5)
    for (int c = 0; c < cy; ++c) {
        h = logaddexp_f(sLa[c][lx] + h, sLb[c][lx]);
    }

    // pass 2: apply and emit exp(log_h) in place over lv buffer (== d_out)
    for (int t = t0; t < t0 + CL; ++t) {
        const size_t idx = chanBase + (size_t)t * DD;
        float f = lf[idx];
        float v = lv_out[idx];
        h = logaddexp_f(h + f, v);
        lv_out[idx] = __expf(h);
    }
}

// ---------------- launch ----------------

extern "C" void kernel_launch(void* const* d_in, const int* in_sizes, int n_in,
                              void* d_out, int out_size, void* d_ws, size_t ws_size,
                              hipStream_t stream) {
    const float* x  = (const float*)d_in[0];
    const float* Wf = (const float*)d_in[1];
    const float* Wi = (const float*)d_in[2];
    const float* Wh = (const float*)d_in[3];
    float* out = (float*)d_out;

    // ws layout: [3*D*D bf16 weights][BT*D bf16 x][BT*D f32 log_f]  (~198 MB)
    char*   ws  = (char*)d_ws;
    __bf16* wbf = (__bf16*)ws;
    __bf16* xbf = (__bf16*)(ws + (size_t)3 * DD * DD * 2);
    float*  lf  = (float*) (ws + (size_t)3 * DD * DD * 2 + (size_t)BT * DD * 2);

    const int nw8 = DD * DD / 8;
    cvt_f32_bf16<<<(nw8 + 255) / 256, 256, 0, stream>>>(Wf, wbf,                        nw8);
    cvt_f32_bf16<<<(nw8 + 255) / 256, 256, 0, stream>>>(Wi, wbf + (size_t)DD * DD,      nw8);
    cvt_f32_bf16<<<(nw8 + 255) / 256, 256, 0, stream>>>(Wh, wbf + (size_t)2 * DD * DD,  nw8);
    const int nx8 = (int)((size_t)BT * DD / 8);
    cvt_f32_bf16<<<(nx8 + 255) / 256, 256, 0, stream>>>(x, xbf, nx8);

    dim3 ggrid(DD / 64, BT / 128);               // (16, 256)
    gates_gemm<<<ggrid, 256, 0, stream>>>(xbf, wbf, lf, out);

    scan_kernel<<<BB * (DD / 32), dim3(32, 8), 0, stream>>>(lf, out);
}
